// CHGNetLineGraphConv_66649302499835
// MI455X (gfx1250) — compile-verified, matrix operand order checked by reading, not run on hardware
//
#include <hip/hip_runtime.h>
#include <hip/hip_bf16.h>

typedef __attribute__((ext_vector_type(16))) __bf16 v16bf;
typedef __attribute__((ext_vector_type(8)))  float  v8f;

union FragBf {
    v16bf v;
    uint4 q[2];
    unsigned short s[16];
};

__device__ __forceinline__ unsigned short f32_to_bf16(float f) {
    unsigned int u = __float_as_uint(f);
    unsigned int r = u + 0x7FFFu + ((u >> 16) & 1u);   // round-to-nearest-even
    return (unsigned short)(r >> 16);
}

__device__ __forceinline__ float softplus2f(float x) {
    // softplus(x) - ln(2), numerically stable
    float ax = fabsf(x);
    return fmaxf(x, 0.0f) + log1pf(__expf(-ax)) - 0.6931471805599453f;
}

__device__ __forceinline__ float sigmoidf(float x) {
    return 1.0f / (1.0f + __expf(-x));
}

// ---------------------------------------------------------------------------
// Zero the aggregation buffer (must be re-zeroed every call: graph replay).
// ---------------------------------------------------------------------------
__global__ void zero_kernel(float* __restrict__ p, int n) {
    int i = blockIdx.x * blockDim.x + threadIdx.x;
    if (i < n) p[i] = 0.0f;
}

// ---------------------------------------------------------------------------
// Pack a [Din x 64] fp32 weight matrix into bf16 WMMA B-fragment order:
//   tile(kt,nt) of 32x16, per lane 16 bf16:
//   lanes 0-15: col = nt*16+lane, K = kt*32 + {0..7, 16..23}
//   lanes 16-31: col = nt*16+(lane-16), K = kt*32 + {8..15, 24..31}
// ---------------------------------------------------------------------------
__global__ void pack_wmma_b_kernel(const float* __restrict__ W,
                                   unsigned short* __restrict__ P, int Din) {
    int i = blockIdx.x * blockDim.x + threadIdx.x;
    int total = (Din / 32) * 4 * 32 * 16;
    if (i >= total) return;
    int e       = i & 15;
    int lane    = (i >> 4) & 31;
    int tileIdx = i >> 9;
    int nt      = tileIdx & 3;     // D/16 == 4
    int kt      = tileIdx >> 2;
    int col     = nt * 16 + (lane & 15);
    int base0   = (lane < 16) ? 0 : 8;
    int k       = kt * 32 + ((e < 8) ? (base0 + e) : (base0 + 16 + (e - 8)));
    P[i] = f32_to_bf16(W[k * 64 + col]);
}

// ---------------------------------------------------------------------------
// wmul[n,d] = (bond_expansion[n] @ Ww + bw)[d] * shared_node_weights[n,d]
// ---------------------------------------------------------------------------
__global__ void wmul_kernel(const float* __restrict__ bond,
                            const float* __restrict__ Ww,
                            const float* __restrict__ bw,
                            const float* __restrict__ sharedw,
                            float* __restrict__ wmul, int N) {
    int i = blockIdx.x * blockDim.x + threadIdx.x;
    if (i >= N * 64) return;
    int n = i >> 6, d = i & 63;
    float acc = bw[d];
    const float* br = bond + (size_t)n * 31;
#pragma unroll
    for (int r = 0; r < 31; ++r) acc += br[r] * Ww[r * 64 + d];
    wmul[i] = acc * sharedw[i];
}

// ---------------------------------------------------------------------------
// Pass 1: per-edge gated MLP (bf16 WMMA), weighting, atomic scatter to agg.
// Block = 128 threads = 4 waves, handles a tile of 16 edges.
// Wave w owns N-tile nt=w (cols 16w..16w+15) and computes BOTH gates.
// ---------------------------------------------------------------------------
__global__ __launch_bounds__(128, 4)
void edge_msg_kernel(const int* __restrict__ src, const int* __restrict__ dst,
                     const float* __restrict__ node,
                     const float* __restrict__ edgef,
                     const float* __restrict__ auxf,
                     const float* __restrict__ bn_l, const float* __restrict__ bn_g,
                     const unsigned short* __restrict__ Pl,
                     const unsigned short* __restrict__ Pg,
                     const float* __restrict__ wmul,
                     float* __restrict__ agg, int E) {
    __shared__ unsigned short abuf[16][264];   // 256 + 8 pad -> conflict-free
    __shared__ int sSrc[16], sDst[16];

    const int tid = threadIdx.x;
    const int eb  = blockIdx.x * 16;

    if (tid < 16) {
        int e  = eb + tid;
        int ec = (e < E) ? e : (E - 1);
        sSrc[tid] = src[ec];
        sDst[tid] = dst[ec];
    }
    __syncthreads();

    // Gather concat(node[src] | edge | aux | node[dst]) -> bf16 in LDS.
    {
        int er  = tid >> 3;        // edge row in tile
        int seg = tid & 7;         // 8 segments of 32 cols
        int eg  = eb + er; if (eg >= E) eg = E - 1;
        const float* base;
        if (seg < 2)      base = node  + (size_t)sSrc[er] * 64 + seg * 32;
        else if (seg < 4) base = edgef + (size_t)eg * 64 + (seg - 2) * 32;
        else if (seg < 6) base = auxf  + (size_t)eg * 64 + (seg - 4) * 32;
        else              base = node  + (size_t)sDst[er] * 64 + (seg - 6) * 32;
#pragma unroll
        for (int j = 0; j < 32; j += 4) {
            float4 v = *(const float4*)(base + j);
            int c = seg * 32 + j;
            abuf[er][c + 0] = f32_to_bf16(v.x);
            abuf[er][c + 1] = f32_to_bf16(v.y);
            abuf[er][c + 2] = f32_to_bf16(v.z);
            abuf[er][c + 3] = f32_to_bf16(v.w);
        }
    }
    __syncthreads();

    const int lane  = tid & 31;
    const int nt    = tid >> 5;
    const int col   = nt * 16 + (lane & 15);
    const int base0 = (lane < 16) ? 0 : 8;
    const int r     = lane & 15;

    v8f accl = {}; v8f accg = {};
#pragma unroll
    for (int kt = 0; kt < 8; ++kt) {
        FragBf a, bl, bg;
        a.q[0] = *(const uint4*)&abuf[r][kt * 32 + base0];
        a.q[1] = *(const uint4*)&abuf[r][kt * 32 + base0 + 16];
        const uint4* pl = (const uint4*)Pl + ((size_t)(kt * 4 + nt) * 32 + lane) * 2;
        const uint4* pg = (const uint4*)Pg + ((size_t)(kt * 4 + nt) * 32 + lane) * 2;
        bl.q[0] = pl[0]; bl.q[1] = pl[1];
        bg.q[0] = pg[0]; bg.q[1] = pg[1];
        accl = __builtin_amdgcn_wmma_f32_16x16x32_bf16(false, a.v, false, bl.v,
                                                       (short)0, accl, false, false);
        accg = __builtin_amdgcn_wmma_f32_16x16x32_bf16(false, a.v, false, bg.v,
                                                       (short)0, accg, false, false);
    }

    const float biasl = bn_l[col];
    const float biasg = bn_g[col];
    const int mbase = (lane < 16) ? 0 : 8;
#pragma unroll
    for (int i = 0; i < 8; ++i) {
        int m = mbase + i;
        int e = eb + m;
        if (e < E) {
            float msg = softplus2f(accl[i] + biasl) * sigmoidf(accg[i] + biasg);
            int s = sSrc[m], d = sDst[m];
            msg *= wmul[(size_t)s * 64 + col] * wmul[(size_t)d * 64 + col];
            atomicAdd(&agg[(size_t)d * 64 + col], msg);
        }
    }
}

// ---------------------------------------------------------------------------
// new_node = node + agg @ Wo   (16-node tiles, bf16 WMMA, K=64)
// ---------------------------------------------------------------------------
__global__ __launch_bounds__(128, 4)
void node_update_kernel(const float* __restrict__ node,
                        const float* __restrict__ agg,
                        const unsigned short* __restrict__ Po,
                        float* __restrict__ out, int N) {
    const int tid   = threadIdx.x;
    const int nb    = blockIdx.x * 16;
    const int lane  = tid & 31;
    const int nt    = tid >> 5;
    const int col   = nt * 16 + (lane & 15);
    const int base0 = (lane < 16) ? 0 : 8;
    int r = nb + (lane & 15); if (r >= N) r = N - 1;

    v8f acc = {};
#pragma unroll
    for (int kt = 0; kt < 2; ++kt) {
        FragBf a, b;
        const float* ap = agg + (size_t)r * 64 + kt * 32 + base0;
        float4 f0 = *(const float4*)(ap);
        float4 f1 = *(const float4*)(ap + 4);
        float4 f2 = *(const float4*)(ap + 16);
        float4 f3 = *(const float4*)(ap + 20);
        a.s[0]  = f32_to_bf16(f0.x); a.s[1]  = f32_to_bf16(f0.y);
        a.s[2]  = f32_to_bf16(f0.z); a.s[3]  = f32_to_bf16(f0.w);
        a.s[4]  = f32_to_bf16(f1.x); a.s[5]  = f32_to_bf16(f1.y);
        a.s[6]  = f32_to_bf16(f1.z); a.s[7]  = f32_to_bf16(f1.w);
        a.s[8]  = f32_to_bf16(f2.x); a.s[9]  = f32_to_bf16(f2.y);
        a.s[10] = f32_to_bf16(f2.z); a.s[11] = f32_to_bf16(f2.w);
        a.s[12] = f32_to_bf16(f3.x); a.s[13] = f32_to_bf16(f3.y);
        a.s[14] = f32_to_bf16(f3.z); a.s[15] = f32_to_bf16(f3.w);
        const uint4* pb = (const uint4*)Po + ((size_t)(kt * 4 + nt) * 32 + lane) * 2;
        b.q[0] = pb[0]; b.q[1] = pb[1];
        acc = __builtin_amdgcn_wmma_f32_16x16x32_bf16(false, a.v, false, b.v,
                                                      (short)0, acc, false, false);
    }

    const int mbase = (lane < 16) ? 0 : 8;
#pragma unroll
    for (int i = 0; i < 8; ++i) {
        int n = nb + mbase + i;
        if (n < N)
            out[(size_t)n * 64 + col] = node[(size_t)n * 64 + col] + acc[i];
    }
}

// ---------------------------------------------------------------------------
// Pass 2: edge (angle) update using new_node; residual add into output.
// ---------------------------------------------------------------------------
__global__ __launch_bounds__(128, 4)
void edge_update_kernel(const int* __restrict__ src, const int* __restrict__ dst,
                        const float* __restrict__ newnode,
                        const float* __restrict__ edgef,
                        const float* __restrict__ auxf,
                        const float* __restrict__ be_l, const float* __restrict__ be_g,
                        const unsigned short* __restrict__ Pl,
                        const unsigned short* __restrict__ Pg,
                        float* __restrict__ out_edge, int E) {
    __shared__ unsigned short abuf[16][264];
    __shared__ int sSrc[16], sDst[16];

    const int tid = threadIdx.x;
    const int eb  = blockIdx.x * 16;

    if (tid < 16) {
        int e  = eb + tid;
        int ec = (e < E) ? e : (E - 1);
        sSrc[tid] = src[ec];
        sDst[tid] = dst[ec];
    }
    __syncthreads();

    {
        int er  = tid >> 3;
        int seg = tid & 7;
        int eg  = eb + er; if (eg >= E) eg = E - 1;
        const float* base;
        if (seg < 2)      base = newnode + (size_t)sSrc[er] * 64 + seg * 32;
        else if (seg < 4) base = edgef   + (size_t)eg * 64 + (seg - 2) * 32;
        else if (seg < 6) base = auxf    + (size_t)eg * 64 + (seg - 4) * 32;
        else              base = newnode + (size_t)sDst[er] * 64 + (seg - 6) * 32;
#pragma unroll
        for (int j = 0; j < 32; j += 4) {
            float4 v = *(const float4*)(base + j);
            int c = seg * 32 + j;
            abuf[er][c + 0] = f32_to_bf16(v.x);
            abuf[er][c + 1] = f32_to_bf16(v.y);
            abuf[er][c + 2] = f32_to_bf16(v.z);
            abuf[er][c + 3] = f32_to_bf16(v.w);
        }
    }
    __syncthreads();

    const int lane  = tid & 31;
    const int nt    = tid >> 5;
    const int col   = nt * 16 + (lane & 15);
    const int base0 = (lane < 16) ? 0 : 8;
    const int r     = lane & 15;

    v8f accl = {}; v8f accg = {};
#pragma unroll
    for (int kt = 0; kt < 8; ++kt) {
        FragBf a, bl, bg;
        a.q[0] = *(const uint4*)&abuf[r][kt * 32 + base0];
        a.q[1] = *(const uint4*)&abuf[r][kt * 32 + base0 + 16];
        const uint4* pl = (const uint4*)Pl + ((size_t)(kt * 4 + nt) * 32 + lane) * 2;
        const uint4* pg = (const uint4*)Pg + ((size_t)(kt * 4 + nt) * 32 + lane) * 2;
        bl.q[0] = pl[0]; bl.q[1] = pl[1];
        bg.q[0] = pg[0]; bg.q[1] = pg[1];
        accl = __builtin_amdgcn_wmma_f32_16x16x32_bf16(false, a.v, false, bl.v,
                                                       (short)0, accl, false, false);
        accg = __builtin_amdgcn_wmma_f32_16x16x32_bf16(false, a.v, false, bg.v,
                                                       (short)0, accg, false, false);
    }

    const float biasl = be_l[col];
    const float biasg = be_g[col];
    const int mbase = (lane < 16) ? 0 : 8;
#pragma unroll
    for (int i = 0; i < 8; ++i) {
        int m = mbase + i;
        int e = eb + m;
        if (e < E) {
            float upd = softplus2f(accl[i] + biasl) * sigmoidf(accg[i] + biasg);
            out_edge[(size_t)e * 64 + col] = edgef[(size_t)e * 64 + col] + upd;
        }
    }
}

// ---------------------------------------------------------------------------
extern "C" void kernel_launch(void* const* d_in, const int* in_sizes, int n_in,
                              void* d_out, int out_size, void* d_ws, size_t ws_size,
                              hipStream_t stream) {
    const int*   src     = (const int*)d_in[0];
    const int*   dst     = (const int*)d_in[1];
    const float* node    = (const float*)d_in[2];
    const float* edgef   = (const float*)d_in[3];
    const float* auxf    = (const float*)d_in[4];
    const float* bond    = (const float*)d_in[5];
    const float* sharedw = (const float*)d_in[6];
    const float* Wn_l    = (const float*)d_in[7];
    const float* bn_l    = (const float*)d_in[8];
    const float* Wn_g    = (const float*)d_in[9];
    const float* bn_g    = (const float*)d_in[10];
    const float* Wo      = (const float*)d_in[11];
    const float* Ww      = (const float*)d_in[12];
    const float* bw      = (const float*)d_in[13];
    const float* We_l    = (const float*)d_in[14];
    const float* be_l    = (const float*)d_in[15];
    const float* We_g    = (const float*)d_in[16];
    const float* be_g    = (const float*)d_in[17];

    const int E = in_sizes[0];
    const int N = in_sizes[2] / 64;

    float* out_node = (float*)d_out;
    float* out_edge = out_node + (size_t)N * 64;

    // Workspace layout
    char*  ws   = (char*)d_ws;
    size_t off  = 0;
    float* wmul = (float*)(ws + off); off += (size_t)N * 64 * 4;
    float* agg  = (float*)(ws + off); off += (size_t)N * 64 * 4;
    unsigned short* Pnl = (unsigned short*)(ws + off); off += 16384 * 2;
    unsigned short* Png = (unsigned short*)(ws + off); off += 16384 * 2;
    unsigned short* Pel = (unsigned short*)(ws + off); off += 16384 * 2;
    unsigned short* Peg = (unsigned short*)(ws + off); off += 16384 * 2;
    unsigned short* Po  = (unsigned short*)(ws + off); off += 4096 * 2;

    // 1. zero aggregation buffer (deterministic per call)
    {
        int n = N * 64;
        zero_kernel<<<(n + 255) / 256, 256, 0, stream>>>(agg, n);
    }
    // 2. pack weights to bf16 WMMA B-fragment layout (tiny; L2-resident after)
    pack_wmma_b_kernel<<<(16384 + 255) / 256, 256, 0, stream>>>(Wn_l, Pnl, 256);
    pack_wmma_b_kernel<<<(16384 + 255) / 256, 256, 0, stream>>>(Wn_g, Png, 256);
    pack_wmma_b_kernel<<<(16384 + 255) / 256, 256, 0, stream>>>(We_l, Pel, 256);
    pack_wmma_b_kernel<<<(16384 + 255) / 256, 256, 0, stream>>>(We_g, Peg, 256);
    pack_wmma_b_kernel<<<(4096 + 255) / 256, 256, 0, stream>>>(Wo, Po, 64);
    // 3. per-node weights: (bond @ Ww + bw) * shared
    {
        int n = N * 64;
        wmul_kernel<<<(n + 255) / 256, 256, 0, stream>>>(bond, Ww, bw, sharedw, wmul, N);
    }
    // 4. pass 1: edge messages + scatter
    edge_msg_kernel<<<(E + 15) / 16, 128, 0, stream>>>(
        src, dst, node, edgef, auxf, bn_l, bn_g, Pnl, Png, wmul, agg, E);
    // 5. node update: new_node = node + agg @ Wo
    node_update_kernel<<<(N + 15) / 16, 128, 0, stream>>>(node, agg, Po, out_node, N);
    // 6. pass 2: edge update using new_node
    edge_update_kernel<<<(E + 15) / 16, 128, 0, stream>>>(
        src, dst, out_node, edgef, auxf, be_l, be_g, Pel, Peg, out_edge, E);
}
